// StarE_PyG_Encoder_54589034332743
// MI455X (gfx1250) — matrix operand order repaired and checked by reading, not compiled, not run on hardware
//
#include <hip/hip_runtime.h>
#include <hip/hip_bf16.h>

// ---------------------------------------------------------------------------
// StarE encoder for MI455X (gfx1250, wave32).
// Heavy GEMMs run through v_wmma_f32_16x16x32_bf16 (fp32 accumulate).
// ---------------------------------------------------------------------------

typedef __attribute__((ext_vector_type(16))) __bf16 v16bf;
typedef __attribute__((ext_vector_type(8)))  float  v8f;

#define GEMM_BM 128
#define GEMM_BN 32
#define GEMM_BK 32

#define EPI_STORE     0
#define EPI_SCATTER   1
#define EPI_BIAS_RELU 2
#define EPI_BIAS      3

// Packed f32x2 -> bf16x2 (single v_cvt_pk_bf16_f32; low half = first arg).
static __device__ __forceinline__ unsigned pack2_bf16(float a, float b)
{
    __hip_bfloat162 t = __float22bfloat162_rn(make_float2(a, b));
    unsigned r;
    __builtin_memcpy(&r, &t, sizeof(r));
    return r;
}

// ---------------------------------------------------------------------------
// Generic bf16-WMMA GEMM: C[M,Nn] = A[M,K] * B (B optionally transposed).
// A/B are f32 in memory; converted to bf16 while staging into LDS.
// 8 waves per block; block tile 128x32; each wave computes 16x32 via two
// WMMAs sharing one A fragment. Staging and fragment loads are branch-free
// in the steady state and fully vectorized (float4 global loads, b128/b64
// LDS traffic, packed bf16 conversion).
// ---------------------------------------------------------------------------
__global__ __launch_bounds__(256)
void gemm_bf16_wmma(const float* __restrict__ A, int lda,
                    const float* __restrict__ B, int ldb, int b_trans,
                    float* __restrict__ C, int ldc,
                    const int* __restrict__ rowidx,
                    const float* __restrict__ bias,
                    int M, int K, int Nn, int mode)
{
    // Row stride 40 bf16 = 80 bytes: 16B-aligned rows, conflict-free b128.
    __shared__ __bf16 As[GEMM_BM][40];   // [m][k]
    __shared__ __bf16 Bs[GEMM_BN][40];   // [n][k]  (transposed in LDS)

    const int tid  = threadIdx.x;
    const int wave = tid >> 5;
    const int lane = tid & 31;
    const int gm0  = blockIdx.x * GEMM_BM;
    const int gn0  = blockIdx.y * GEMM_BN;
    const int lm   = lane & 15;   // M (A frag) / N (B frag, C frag)
    const int lk   = lane >> 4;   // half-select

    v8f acc0 = {0.f, 0.f, 0.f, 0.f, 0.f, 0.f, 0.f, 0.f};
    v8f acc1 = {0.f, 0.f, 0.f, 0.f, 0.f, 0.f, 0.f, 0.f};

    // A staging: thread -> (row, 16-k chunk). Row index clamped (results for
    // clamped rows are masked in the epilogue, so garbage is harmless).
    const int ar  = tid >> 1;             // 0..127
    const int ak0 = (tid & 1) * 16;       // 0 or 16
    const int agr = (gm0 + ar < M) ? (gm0 + ar) : (M - 1);
    const float* __restrict__ Arow = A + (size_t)agr * lda;

    // B staging indexing (two layouts, both 4 elements/thread):
    //   b_trans: thread -> (n-row, 4-k chunk); row-contiguous in LDS.
    //   !b_trans: thread -> (k-row, 4-n chunk); transposed scatter into LDS.
    const int bq  = tid >> 3;             // 0..31 (n-row or k-row)
    const int b4  = (tid & 7) << 2;       // 0,4,...,28 (k or n sub-offset)
    const int bnc = (gn0 + bq < Nn) ? (gn0 + bq) : (Nn - 1);
    const float* __restrict__ BrowT = B + (size_t)bnc * ldb;   // b_trans rows
    const bool n_full = (gn0 + GEMM_BN) <= Nn;

    for (int k0 = 0; k0 < K; k0 += GEMM_BK) {
        const bool k_full = (k0 + GEMM_BK) <= K;

        // ---- stage A: 16 consecutive k per thread ----
        union { __bf16 h[16]; unsigned u[8]; uint4 q[2]; } pk;
        if (k_full) {
            const float4* __restrict__ ap = (const float4*)(Arow + k0 + ak0);
            float4 f0 = ap[0], f1 = ap[1], f2 = ap[2], f3 = ap[3];
            pk.u[0] = pack2_bf16(f0.x, f0.y);
            pk.u[1] = pack2_bf16(f0.z, f0.w);
            pk.u[2] = pack2_bf16(f1.x, f1.y);
            pk.u[3] = pack2_bf16(f1.z, f1.w);
            pk.u[4] = pack2_bf16(f2.x, f2.y);
            pk.u[5] = pack2_bf16(f2.z, f2.w);
            pk.u[6] = pack2_bf16(f3.x, f3.y);
            pk.u[7] = pack2_bf16(f3.z, f3.w);
        } else {
            // K-tail: clamped address + select (branch-free)
#pragma unroll
            for (int u = 0; u < 16; ++u) {
                const int kk = k0 + ak0 + u;
                const int kc = kk < K ? kk : K - 1;
                const float v = Arow[kc];
                pk.h[u] = (__bf16)(kk < K ? v : 0.f);
            }
        }
        {
            uint4* dst = (uint4*)&As[ar][ak0];
            dst[0] = pk.q[0];
            dst[1] = pk.q[1];
        }

        // ---- stage B: 32x32 tile, 4 elements/thread, vectorized ----
        if (b_trans) {
            union { __bf16 h[4]; unsigned u[2]; uint2 q; } pb;
            if (k_full) {
                float4 f = *(const float4*)(BrowT + k0 + b4);
                pb.u[0] = pack2_bf16(f.x, f.y);
                pb.u[1] = pack2_bf16(f.z, f.w);
            } else {
#pragma unroll
                for (int u = 0; u < 4; ++u) {
                    const int kk = k0 + b4 + u;
                    const int kc = kk < K ? kk : K - 1;
                    const float v = BrowT[kc];
                    pb.h[u] = (__bf16)(kk < K ? v : 0.f);
                }
            }
            *(uint2*)&Bs[bq][b4] = pb.q;
        } else {
            const int gk  = k0 + bq;
            const int gkc = gk < K ? gk : K - 1;
            const float* __restrict__ Brow = B + (size_t)gkc * ldb;
            float bv[4];
            if (n_full) {
                float4 f = *(const float4*)(Brow + gn0 + b4);
                bv[0] = f.x; bv[1] = f.y; bv[2] = f.z; bv[3] = f.w;
            } else {
#pragma unroll
                for (int u = 0; u < 4; ++u) {
                    const int gn = gn0 + b4 + u;
                    bv[u] = Brow[gn < Nn ? gn : Nn - 1];
                }
            }
            const bool kin = gk < K;
#pragma unroll
            for (int u = 0; u < 4; ++u)
                Bs[b4 + u][bq] = (__bf16)(kin ? bv[u] : 0.f);
        }
        __syncthreads();

        // ---- fragments: 2x b128 LDS loads each (contiguous k = 16*lk + s, same order for A and B) ----
        union { v16bf v; uint4 q[2]; } af, bf0, bf1;
        {
            const uint4* ap = (const uint4*)&As[wave * 16 + lm][0];
            af.q[0] = ap[lk * 2];        // k = lk*16 .. lk*16+7
            af.q[1] = ap[lk * 2 + 1];    // k = lk*16+8 .. lk*16+15
            const uint4* bp0 = (const uint4*)&Bs[lm][0];
            bf0.q[0] = bp0[lk * 2];      // k = lk*16 .. lk*16+7
            bf0.q[1] = bp0[lk * 2 + 1];  // k = lk*16+8 .. lk*16+15
            const uint4* bp1 = (const uint4*)&Bs[16 + lm][0];
            bf1.q[0] = bp1[lk * 2];
            bf1.q[1] = bp1[lk * 2 + 1];
        }
        acc0 = __builtin_amdgcn_wmma_f32_16x16x32_bf16(
                   false, af.v, false, bf0.v, (short)0, acc0, false, false);
        acc1 = __builtin_amdgcn_wmma_f32_16x16x32_bf16(
                   false, af.v, false, bf1.v, (short)0, acc1, false, false);
        __syncthreads();
    }

    // Epilogue. C-layout: lane lm = n, row m = lk*8 + v.
#pragma unroll
    for (int v = 0; v < 8; ++v) {
        const int gm = gm0 + wave * 16 + lk * 8 + v;
        if (gm >= M) continue;
        // N-half 0
        {
            const int gn = gn0 + lm;
            if (gn < Nn) {
                float val = acc0[v];
                if (mode == EPI_STORE) {
                    C[(size_t)gm * ldc + gn] = val;
                } else if (mode == EPI_SCATTER) {
                    atomicAdd(&C[(size_t)rowidx[gm] * ldc + gn], val);
                } else if (mode == EPI_BIAS_RELU) {
                    val += bias[gn];
                    C[(size_t)gm * ldc + gn] = val > 0.f ? val : 0.f;
                } else {
                    C[(size_t)gm * ldc + gn] = val + bias[gn];
                }
            }
        }
        // N-half 1
        {
            const int gn = gn0 + 16 + lm;
            if (gn < Nn) {
                float val = acc1[v];
                if (mode == EPI_STORE) {
                    C[(size_t)gm * ldc + gn] = val;
                } else if (mode == EPI_SCATTER) {
                    atomicAdd(&C[(size_t)rowidx[gm] * ldc + gn], val);
                } else if (mode == EPI_BIAS_RELU) {
                    val += bias[gn];
                    C[(size_t)gm * ldc + gn] = val > 0.f ? val : 0.f;
                } else {
                    C[(size_t)gm * ldc + gn] = val + bias[gn];
                }
            }
        }
    }
}

// ---------------------------------------------------------------------------
// Elementwise / irregular kernels (D = 200, half-dim = 100 hardcoded).
// ---------------------------------------------------------------------------
__global__ void zero_f32(float* p, long n)
{
    long i = (long)blockIdx.x * blockDim.x + threadIdx.x;
    if (i < n) p[i] = 0.f;
}

__global__ void copy_f32(float* dst, const float* src, long n)
{
    long i = (long)blockIdx.x * blockDim.x + threadIdx.x;
    if (i < n) dst[i] = src[i];
}

__global__ void copy_cols(float* dst, int dld, int dcol,
                          const float* src, int sld, int scol,
                          int rows, int cols)
{
    long i = (long)blockIdx.x * blockDim.x + threadIdx.x;
    if (i >= (long)rows * cols) return;
    int r = (int)(i / cols), c = (int)(i % cols);
    dst[(size_t)r * dld + dcol + c] = src[(size_t)r * sld + scol + c];
}

__global__ void deg_accum(const int* __restrict__ rows, float* deg, int E)
{
    long i = (long)blockIdx.x * blockDim.x + threadIdx.x;
    if (i < E) atomicAdd(&deg[rows[i]], 1.f);
}

__global__ void deg_finalize(float* deg, int n)
{
    long i = (long)blockIdx.x * blockDim.x + threadIdx.x;
    if (i < n) {
        float d = deg[i];
        deg[i] = d > 0.f ? rsqrtf(d) : 0.f;
    }
}

// q_accum[qx[q]] += rotate(x[qe[q]], rel_all[qr[q]])
__global__ void qual_rotate_scatter(const float* __restrict__ x,
                                    const float* __restrict__ rel_all,
                                    const int* __restrict__ qr,
                                    const int* __restrict__ qe,
                                    const int* __restrict__ qx,
                                    float* __restrict__ q_accum, int Q)
{
    long i = (long)blockIdx.x * blockDim.x + threadIdx.x;
    if (i >= (long)Q * 100) return;
    int q = (int)(i / 100), j = (int)(i % 100);
    const float* h = x + (size_t)qe[q] * 200;
    const float* r = rel_all + (size_t)qr[q] * 200;
    float hr = h[j], hi = h[j + 100], rr = r[j], ri = r[j + 100];
    float* dst = q_accum + (size_t)qx[q] * 200;
    atomicAdd(&dst[j],       hr * rr - hi * ri);
    atomicAdd(&dst[j + 100], hr * ri + hi * rr);
}

// A2[e] = rotate(x[col[e]], 0.8*rel_all[etype+off] + 0.2*q_edge[e]) * norm[e]
__global__ void edge_rotate_norm(const float* __restrict__ x,
                                 const float* __restrict__ rel_all,
                                 const float* __restrict__ q_edge,
                                 const int* __restrict__ erow,
                                 const int* __restrict__ ecol,
                                 const int* __restrict__ etype,
                                 int rel_off,
                                 const float* __restrict__ dinv,
                                 float* __restrict__ A2, int E)
{
    long i = (long)blockIdx.x * blockDim.x + threadIdx.x;
    if (i >= (long)E * 100) return;
    int e = (int)(i / 100), j = (int)(i % 100);
    float norm = dinv[erow[e]] * dinv[ecol[e]];
    const float* ra = rel_all + (size_t)(etype[e] + rel_off) * 200;
    const float* qv = q_edge + (size_t)e * 200;
    float rr = 0.8f * ra[j]       + 0.2f * qv[j];
    float ri = 0.8f * ra[j + 100] + 0.2f * qv[j + 100];
    const float* h = x + (size_t)ecol[e] * 200;
    float hr = h[j], hi = h[j + 100];
    A2[(size_t)e * 200 + j]       = (hr * rr - hi * ri) * norm;
    A2[(size_t)e * 200 + j + 100] = (hr * ri + hi * rr) * norm;
}

// A3[n] = rotate(x[n], loop_rel)
__global__ void loop_rotate(const float* __restrict__ x,
                            const float* __restrict__ relrow,
                            float* __restrict__ A3, int N)
{
    long i = (long)blockIdx.x * blockDim.x + threadIdx.x;
    if (i >= (long)N * 100) return;
    int n = (int)(i / 100), j = (int)(i % 100);
    float rr = relrow[j], ri = relrow[j + 100];
    const float* h = x + (size_t)n * 200;
    float hr = h[j], hi = h[j + 100];
    A3[(size_t)n * 200 + j]       = hr * rr - hi * ri;
    A3[(size_t)n * 200 + j + 100] = hr * ri + hi * rr;
}

// Per-column mean/var of (v * scale). One block per column.
__global__ void bn_colstats(const float* __restrict__ v, float scale,
                            float* __restrict__ mu, float* __restrict__ var,
                            int rows, int cols)
{
    __shared__ float ss[256], ss2[256];
    const int c = blockIdx.x;
    float s = 0.f, s2 = 0.f;
    for (int r = threadIdx.x; r < rows; r += blockDim.x) {
        float t = v[(size_t)r * cols + c] * scale;
        s += t; s2 += t * t;
    }
    ss[threadIdx.x] = s; ss2[threadIdx.x] = s2;
    __syncthreads();
    for (int st = 128; st > 0; st >>= 1) {
        if (threadIdx.x < st) {
            ss[threadIdx.x]  += ss[threadIdx.x + st];
            ss2[threadIdx.x] += ss2[threadIdx.x + st];
        }
        __syncthreads();
    }
    if (threadIdx.x == 0) {
        float m = ss[0] / rows;
        mu[c]  = m;
        var[c] = ss2[0] / rows - m * m;
    }
}

__global__ void bn_apply(const float* __restrict__ v,
                         const float* __restrict__ mu,
                         const float* __restrict__ var,
                         const float* __restrict__ g,
                         const float* __restrict__ b,
                         float* __restrict__ out,
                         float scale, int post_relu, int rows, int cols)
{
    long i = (long)blockIdx.x * blockDim.x + threadIdx.x;
    if (i >= (long)rows * cols) return;
    int c = (int)(i % cols);
    float t = v[i] * scale;
    float o = g[c] * (t - mu[c]) * rsqrtf(var[c] + 1e-5f) + b[c];
    if (post_relu && o < 0.f) o = 0.f;
    out[i] = o;
}

// Column sums of t's first `gridDim.x` columns (t has ld=200).
__global__ void col_sum(const float* __restrict__ t, float* __restrict__ csum,
                        int rows)
{
    __shared__ float ss[256];
    const int c = blockIdx.x;
    float s = 0.f;
    for (int r = threadIdx.x; r < rows; r += blockDim.x)
        s += t[(size_t)r * 200 + c];
    ss[threadIdx.x] = s;
    __syncthreads();
    for (int st = 128; st > 0; st >>= 1) {
        if (threadIdx.x < st) ss[threadIdx.x] += ss[threadIdx.x + st];
        __syncthreads();
    }
    if (threadIdx.x == 0) csum[c] = ss[0];
}

// nf = dot(colsum(U), colsum(V)) / N + 1e-6
__global__ void compute_nf(const float* __restrict__ csum, float* nf, int rows)
{
    __shared__ float ss[64];
    int i = threadIdx.x;
    ss[i] = (i < 50) ? csum[i] * csum[50 + i] : 0.f;
    __syncthreads();
    for (int st = 32; st > 0; st >>= 1) {
        if (i < st) ss[i] += ss[i + st];
        __syncthreads();
    }
    if (i == 0) nf[0] = ss[0] / rows + 1e-6f;
}

// VtZ[a,b] = sum_n V[n,a] * Z[n,b]; one block per (a,b).
__global__ void vtz_kernel(const float* __restrict__ t, float* __restrict__ VtZ,
                           int rows)
{
    __shared__ float ss[256];
    const int a = blockIdx.x / 50, b = blockIdx.x % 50;
    float s = 0.f;
    for (int r = threadIdx.x; r < rows; r += blockDim.x)
        s += t[(size_t)r * 200 + 50 + a] * t[(size_t)r * 200 + 100 + b];
    ss[threadIdx.x] = s;
    __syncthreads();
    for (int st = 128; st > 0; st >>= 1) {
        if (threadIdx.x < st) ss[threadIdx.x] += ss[threadIdx.x + st];
        __syncthreads();
    }
    if (threadIdx.x == 0) VtZ[a * 50 + b] = ss[0];
}

// h[:, 0:50] = U @ VtZ / nf ; h[:, 50:100] = T   (h has ld=500)
__global__ void xglobal_kernel(const float* __restrict__ t,
                               const float* __restrict__ VtZ,
                               const float* __restrict__ nf,
                               float* __restrict__ h, int N)
{
    long i = (long)blockIdx.x * blockDim.x + threadIdx.x;
    if (i >= (long)N * 100) return;
    int n = (int)(i / 100), b = (int)(i % 100);
    if (b < 50) {
        float s = 0.f;
#pragma unroll
        for (int a = 0; a < 50; ++a)
            s += t[(size_t)n * 200 + a] * VtZ[a * 50 + b];
        h[(size_t)n * 500 + b] = s / nf[0];
    } else {
        h[(size_t)n * 500 + b] = t[(size_t)n * 200 + 150 + (b - 50)];
    }
}

// ---------------------------------------------------------------------------
// Host orchestration
// ---------------------------------------------------------------------------
extern "C" void kernel_launch(void* const* d_in, const int* in_sizes, int n_in,
                              void* d_out, int out_size, void* d_ws, size_t ws_size,
                              hipStream_t stream)
{
    (void)in_sizes; (void)n_in; (void)out_size; (void)ws_size;
    const int N = 30000, E = 250000, Q = 150000, D = 200, R = 400, L = 2;
    const int K4 = 200;          // 4*K
    const int NREL = 2 * R + 1;  // 801

    const int*   edge_index = (const int*)d_in[0];   // (2,E)
    const int*   edge_type  = (const int*)d_in[1];
    const int*   quals_rel  = (const int*)d_in[2];
    const int*   quals_ent  = (const int*)d_in[3];
    const int*   quals_edge = (const int*)d_in[4];
    const float* ent_emb    = (const float*)d_in[5];
    const float* init_rel   = (const float*)d_in[6];
    const float* loop_rel   = (const float*)d_in[7];
    const float* w_in       = (const float*)d_in[8];
    const float* w_out      = (const float*)d_in[9];
    const float* w_loop     = (const float*)d_in[10];
    const float* w_rel      = (const float*)d_in[11];
    const float* w_q        = (const float*)d_in[12];
    const float* conv_gamma = (const float*)d_in[13];
    const float* conv_beta  = (const float*)d_in[14];
    const float* lrga_w     = (const float*)d_in[15];
    const float* lrga_b     = (const float*)d_in[16];
    const float* dimred_w   = (const float*)d_in[17]; // (L,200,500)
    const float* dimred_b   = (const float*)d_in[18];
    const float* fuse_gamma = (const float*)d_in[19];
    const float* fuse_beta  = (const float*)d_in[20];

    float* out_x = (float*)d_out;                 // N*D
    float* out_r = out_x + (size_t)N * D;         // 801*D

    // --- workspace carve (q_accum aliased with rotated-A buffer A2) ---
    size_t off = 0;
    auto carve = [&](size_t nbytes) -> float* {
        float* p = (float*)((char*)d_ws + off);
        off += (nbytes + 255) & ~(size_t)255;
        return p;
    };
    float* q_accum = carve((size_t)E * D * 4);   // also reused as A2/A3
    float* q_edge  = carve((size_t)E * D * 4);
    float* x_sum   = carve((size_t)N * D * 4);
    float* x_local = carve((size_t)N * D * 4);
    float* xbuf    = carve((size_t)N * D * 4);
    float* tbuf    = carve((size_t)N * K4 * 4);
    float* hbuf    = carve((size_t)N * 500 * 4);
    float* rel_all = carve((size_t)(NREL + 1) * D * 4);  // 802 rows
    float* rel_tmp = carve((size_t)NREL * D * 4);
    float* dinv_in = carve((size_t)N * 4);
    float* dinv_out= carve((size_t)N * 4);
    float* mu      = carve(D * 4);
    float* var     = carve(D * 4);
    float* csum    = carve(D * 4);
    float* VtZ     = carve(2500 * 4);
    float* nf      = carve(256);
    float* A2buf   = q_accum;   // alias: q_accum dead once q_edge GEMM finishes

    auto nblk = [](long n) { return (unsigned)((n + 255) / 256); };
    auto launch_gemm = [&](const float* A, int lda, const float* B, int ldb,
                           int btr, float* C, int ldc, const int* rowidx,
                           const float* bias, int M, int Kd, int Nn, int mode) {
        dim3 grid((M + GEMM_BM - 1) / GEMM_BM, (Nn + GEMM_BN - 1) / GEMM_BN);
        gemm_bf16_wmma<<<grid, 256, 0, stream>>>(A, lda, B, ldb, btr, C, ldc,
                                                 rowidx, bias, M, Kd, Nn, mode);
    };

    const int* erow_in  = edge_index;       // edge_index[0]
    const int* ecol_in  = edge_index + E;   // edge_index[1]

    // Degrees (layer-invariant): in-dir rows = ei[0], out-dir rows = ei[1].
    zero_f32<<<nblk(N), 256, 0, stream>>>(dinv_in,  N);
    zero_f32<<<nblk(N), 256, 0, stream>>>(dinv_out, N);
    deg_accum<<<nblk(E), 256, 0, stream>>>(erow_in, dinv_in,  E);
    deg_accum<<<nblk(E), 256, 0, stream>>>(ecol_in, dinv_out, E);
    deg_finalize<<<nblk(N), 256, 0, stream>>>(dinv_in,  N);
    deg_finalize<<<nblk(N), 256, 0, stream>>>(dinv_out, N);

    for (int i = 0; i < L; ++i) {
        const float* x = (i == 0) ? ent_emb : xbuf;
        const float* wq_i    = w_q    + (size_t)i * D * D;
        const float* win_i   = w_in   + (size_t)i * D * D;
        const float* wout_i  = w_out  + (size_t)i * D * D;
        const float* wloop_i = w_loop + (size_t)i * D * D;
        const float* wrel_i  = w_rel  + (size_t)i * D * D;

        // rel_all = concat(rel, loop_rel[i])
        copy_f32<<<nblk((long)NREL * D), 256, 0, stream>>>(
            rel_all, (i == 0) ? init_rel : rel_tmp, (long)NREL * D);
        copy_f32<<<nblk(D), 256, 0, stream>>>(
            rel_all + (size_t)NREL * D, loop_rel + (size_t)i * D, D);

        // --- qualifier aggregation (shared by both directions) ---
        zero_f32<<<nblk((long)E * D), 256, 0, stream>>>(q_accum, (long)E * D);
        qual_rotate_scatter<<<nblk((long)Q * 100), 256, 0, stream>>>(
            x, rel_all, quals_rel, quals_ent, quals_edge, q_accum, Q);
        launch_gemm(q_accum, D, wq_i, D, 0, q_edge, D, nullptr, nullptr,
                    E, D, D, EPI_STORE);

        // --- self-loop term (STORE initializes x_sum) ---
        loop_rotate<<<nblk((long)N * 100), 256, 0, stream>>>(
            x, rel_all + (size_t)NREL * D, A2buf, N);
        launch_gemm(A2buf, D, wloop_i, D, 0, x_sum, D, nullptr, nullptr,
                    N, D, D, EPI_STORE);

        // --- in direction: rows = ei[0], cols = ei[1] ---
        edge_rotate_norm<<<nblk((long)E * 100), 256, 0, stream>>>(
            x, rel_all, q_edge, erow_in, ecol_in, edge_type, 0, dinv_in,
            A2buf, E);
        launch_gemm(A2buf, D, win_i, D, 0, x_sum, D, erow_in, nullptr,
                    E, D, D, EPI_SCATTER);

        // --- out direction: rows = ei[1], cols = ei[0], etype + R ---
        edge_rotate_norm<<<nblk((long)E * 100), 256, 0, stream>>>(
            x, rel_all, q_edge, ecol_in, erow_in, edge_type, R, dinv_out,
            A2buf, E);
        launch_gemm(A2buf, D, wout_i, D, 0, x_sum, D, ecol_in, nullptr,
                    E, D, D, EPI_SCATTER);

        // --- BN((in+out+loop)/3) then ReLU -> x_local ---
        bn_colstats<<<D, 256, 0, stream>>>(x_sum, 1.f / 3.f, mu, var, N, D);
        bn_apply<<<nblk((long)N * D), 256, 0, stream>>>(
            x_sum, mu, var, conv_gamma + (size_t)i * D, conv_beta + (size_t)i * D,
            x_local, 1.f / 3.f, 1, N, D);

        // --- LRGA on layer input x ---
        launch_gemm(x, D, lrga_w + (size_t)i * K4 * D, D, 1, tbuf, K4,
                    nullptr, lrga_b + (size_t)i * K4, N, D, K4, EPI_BIAS_RELU);
        col_sum<<<100, 256, 0, stream>>>(tbuf, csum, N);      // colsum U|V
        compute_nf<<<1, 64, 0, stream>>>(csum, nf, N);
        vtz_kernel<<<2500, 256, 0, stream>>>(tbuf, VtZ, N);
        xglobal_kernel<<<nblk((long)N * 100), 256, 0, stream>>>(
            tbuf, VtZ, nf, hbuf, N);

        // h = [x_global | x_local | x]
        copy_cols<<<nblk((long)N * D), 256, 0, stream>>>(
            hbuf, 500, 100, x_local, D, 0, N, D);
        copy_cols<<<nblk((long)N * D), 256, 0, stream>>>(
            hbuf, 500, 300, x, D, 0, N, D);

        // dimension reduction: h @ dimred_w.T + b
        float* hdst = (i < L - 1) ? x_sum : out_x;
        int    hmod = (i < L - 1) ? EPI_BIAS_RELU : EPI_BIAS;
        launch_gemm(hbuf, 500, dimred_w + (size_t)i * D * 500, 500, 1,
                    hdst, D, nullptr, dimred_b + (size_t)i * D,
                    N, 500, D, hmod);
        if (i < L - 1) {
            bn_colstats<<<D, 256, 0, stream>>>(x_sum, 1.f, mu, var, N, D);
            bn_apply<<<nblk((long)N * D), 256, 0, stream>>>(
                x_sum, mu, var, fuse_gamma, fuse_beta, xbuf, 1.f, 0, N, D);
        }

        // relation update: (rel_all @ w_rel)[:-1]
        float* rdst = (i < L - 1) ? rel_tmp : out_r;
        launch_gemm(rel_all, D, wrel_i, D, 0, rdst, D, nullptr, nullptr,
                    NREL, D, D, EPI_STORE);
    }
}
